// RNN_M2M_21208548507903
// MI455X (gfx1250) — compile-verified, matrix operand order checked by reading, not run on hardware
//
#include <hip/hip_runtime.h>
#include <hip/hip_bf16.h>

typedef __attribute__((ext_vector_type(16))) _Float16 v16h;
typedef __attribute__((ext_vector_type(8)))  _Float16 v8h;
typedef __attribute__((ext_vector_type(8)))  float    v8f;

#define B_   256
#define T_   168
#define F_   128
#define H_   512
#define H2_  1024
#define G4_  2048   /* 4*H  */
#define G8_  4096   /* 4*2H */
#define L_   48

union AFrag { v16h v; v8h h[2]; };

__device__ __forceinline__ float sigmoidf_(float x) { return 1.0f / (1.0f + __expf(-x)); }

// ---------------------------------------------------------------------------
// One-time prep kernels
// ---------------------------------------------------------------------------

// input_seq [B,T,F] f32 -> X16 [T,B,F] f16
__global__ void xcvt_kernel(const float* __restrict__ x, _Float16* __restrict__ X16) {
    int idx = blockIdx.x * 256 + threadIdx.x;           // T*B*F
    if (idx >= T_ * B_ * F_) return;
    int f  = idx & (F_ - 1);
    int tb = idx >> 7;
    int b  = tb & (B_ - 1);
    int t  = tb >> 8;
    X16[idx] = (_Float16)x[((size_t)b * T_ + t) * F_ + f];
}

// Pack [4H x K] f32 row-major weights into WMMA B-fragment tile order:
// dst[(kt*Ntiles + nt)*512 + lane*16 + j], lane covers N=lane&15, K-half = lane>>4.
__global__ void pack_w_kernel(const float* __restrict__ Whh, const float* __restrict__ Wih,
                              _Float16* __restrict__ dst, int Ntiles, int Ktiles, int Kh, int Kx) {
    int idx = blockIdx.x * 256 + threadIdx.x;
    int total = Ntiles * Ktiles * 512;
    if (idx >= total) return;
    int tile  = idx >> 9;
    int r     = idx & 511;
    int lane  = r >> 4;
    int j     = r & 15;
    int kt    = tile / Ntiles;
    int nt    = tile - kt * Ntiles;
    int col   = lane & 15;
    int kbase = (lane >> 4) << 3;
    int K     = kt * 32 + ((j < 8) ? (kbase + j) : (16 + kbase + (j - 8)));
    int n     = nt * 16 + col;
    float v   = (K < Kh) ? Whh[(size_t)n * Kh + K] : Wih[(size_t)n * Kx + (K - Kh)];
    dst[idx] = (_Float16)v;
}

__global__ void bias_comb_kernel(const float* bihf, const float* bhhf,
                                 const float* bihb, const float* bhhb,
                                 const float* bihd, const float* bhhd,
                                 float* bcf, float* bcb, float* bcd) {
    int idx = blockIdx.x * 256 + threadIdx.x;          // 2048 + 2048 + 4096
    if (idx < G4_) bcf[idx] = bihf[idx] + bhhf[idx];
    else if (idx < 2 * G4_) { int j = idx - G4_;     bcb[j] = bihb[j] + bhhb[j]; }
    else if (idx < 2 * G4_ + G8_) { int j = idx - 2 * G4_; bcd[j] = bihd[j] + bhhd[j]; }
}

// ---------------------------------------------------------------------------
// Encoder recurrent step: G = [h | x_t] x [Whh | Wih]^T  (K = 640, both dirs)
// 2M x 2N register-blocked wave tile: 4 WMMAs per K-step, A/B fragments reused.
// ---------------------------------------------------------------------------
__global__ void enc_gemm_kernel(const _Float16* __restrict__ h16f, const _Float16* __restrict__ h16b,
                                const _Float16* __restrict__ X16,
                                const _Float16* __restrict__ Wpf, const _Float16* __restrict__ Wpb,
                                float* __restrict__ Gf, float* __restrict__ Gb, int t) {
    const int dir = blockIdx.y;
    const _Float16* h16 = dir ? h16b : h16f;
    const _Float16* Wp  = dir ? Wpb  : Wpf;
    float* G = dir ? Gb : Gf;
    const int teff = dir ? (T_ - 1 - t) : t;

    const int lane = threadIdx.x & 31;
    const int wave = threadIdx.x >> 5;
    const int task = blockIdx.x * 8 + wave;   // 8 (M/2) x 64 (N/2) = 512 tasks
    const int mt0 = (task >> 6) * 2;
    const int nt0 = (task & 63) * 2;

    const int row0 = mt0 * 16 + (lane & 15);
    const int row1 = row0 + 16;
    const int kb   = (lane >> 4) << 3;

    const _Float16* hrow0 = h16 + (size_t)row0 * H_;
    const _Float16* hrow1 = h16 + (size_t)row1 * H_;
    const _Float16* xrow0 = X16 + ((size_t)teff * B_ + row0) * F_;
    const _Float16* xrow1 = X16 + ((size_t)teff * B_ + row1) * F_;

    v8f acc00 = {}, acc01 = {}, acc10 = {}, acc11 = {};
#pragma unroll
    for (int kt = 0; kt < 20; ++kt) {
        const int k0 = kt * 32 + kb;
        const _Float16* ap0 = (kt < 16) ? (hrow0 + k0) : (xrow0 + (k0 - H_));
        const _Float16* ap1 = (kt < 16) ? (hrow1 + k0) : (xrow1 + (k0 - H_));
        AFrag a0, a1;
        a0.h[0] = *(const v8h*)(ap0);
        a0.h[1] = *(const v8h*)(ap0 + 16);
        a1.h[0] = *(const v8h*)(ap1);
        a1.h[1] = *(const v8h*)(ap1 + 16);
        const _Float16* bp = Wp + ((size_t)(kt * 128 + nt0)) * 512 + lane * 16;
        v16h b0 = *(const v16h*)(bp);
        v16h b1 = *(const v16h*)(bp + 512);
        acc00 = __builtin_amdgcn_wmma_f32_16x16x32_f16(false, a0.v, false, b0, (short)0, acc00, false, false);
        acc01 = __builtin_amdgcn_wmma_f32_16x16x32_f16(false, a0.v, false, b1, (short)0, acc01, false, false);
        acc10 = __builtin_amdgcn_wmma_f32_16x16x32_f16(false, a1.v, false, b0, (short)0, acc10, false, false);
        acc11 = __builtin_amdgcn_wmma_f32_16x16x32_f16(false, a1.v, false, b1, (short)0, acc11, false, false);
    }

    const int ncol = lane & 15;
    const int mofs = (lane >> 4) << 3;
    {
        float* g00 = G + (size_t)(mt0 * 16 + mofs) * G4_ + (nt0 * 16 + ncol);
        float* g10 = G + (size_t)((mt0 + 1) * 16 + mofs) * G4_ + (nt0 * 16 + ncol);
#pragma unroll
        for (int v = 0; v < 8; ++v) {
            g00[(size_t)v * G4_]      = acc00[v];
            g00[(size_t)v * G4_ + 16] = acc01[v];
            g10[(size_t)v * G4_]      = acc10[v];
            g10[(size_t)v * G4_ + 16] = acc11[v];
        }
    }
}

__global__ void enc_point_kernel(const float* __restrict__ Gf, const float* __restrict__ Gb,
                                 const float* __restrict__ bcf, const float* __restrict__ bcb,
                                 float* __restrict__ hf, float* __restrict__ cf, _Float16* __restrict__ h16f,
                                 float* __restrict__ hb, float* __restrict__ cb, _Float16* __restrict__ h16b) {
    int idx = blockIdx.x * 256 + threadIdx.x;   // 2*B*H
    int dir = idx >> 17;
    int r   = idx & 131071;
    int b   = r >> 9;
    int j   = r & 511;
    const float* G  = dir ? Gb  : Gf;
    const float* bc = dir ? bcb : bcf;
    float* h = dir ? hb : hf;
    float* c = dir ? cb : cf;
    _Float16* h16 = dir ? h16b : h16f;
    const float* g = G + (size_t)b * G4_;
    float gi = g[j]          + bc[j];
    float gf = g[H_ + j]     + bc[H_ + j];
    float gg = g[2 * H_ + j] + bc[2 * H_ + j];
    float go = g[3 * H_ + j] + bc[3 * H_ + j];
    float cn = sigmoidf_(gf) * c[r] + sigmoidf_(gi) * tanhf(gg);
    c[r] = cn;
    float hn = sigmoidf_(go) * tanhf(cn);
    h[r] = hn;
    h16[r] = (_Float16)hn;
}

// ---------------------------------------------------------------------------
// Decoder
// ---------------------------------------------------------------------------
__global__ void dec_init_kernel(const float* __restrict__ hf, const float* __restrict__ hb,
                                const float* __restrict__ cf, const float* __restrict__ cb,
                                _Float16* __restrict__ hd16, float* __restrict__ cd) {
    int idx = blockIdx.x * 256 + threadIdx.x;   // B*1024
    int b = idx >> 10;
    int j = idx & 1023;
    float hv, cv;
    if (j < H_) { hv = hf[b * H_ + j];        cv = cf[b * H_ + j]; }
    else        { hv = hb[b * H_ + (j - H_)]; cv = cb[b * H_ + (j - H_)]; }
    hd16[idx] = (_Float16)hv;
    cd[idx]   = cv;
}

__global__ void dec_gemm_kernel(const _Float16* __restrict__ hd16, const _Float16* __restrict__ Wpd,
                                float* __restrict__ Gd) {
    const int lane = threadIdx.x & 31;
    const int wave = threadIdx.x >> 5;
    const int task = blockIdx.x * 8 + wave;   // 8 (M/2) x 128 (N/2) = 1024 tasks
    const int mt0 = (task >> 7) * 2;
    const int nt0 = (task & 127) * 2;

    const int row0 = mt0 * 16 + (lane & 15);
    const int kb   = (lane >> 4) << 3;
    const _Float16* hrow0 = hd16 + (size_t)row0 * H2_;
    const _Float16* hrow1 = hrow0 + (size_t)16 * H2_;

    v8f acc00 = {}, acc01 = {}, acc10 = {}, acc11 = {};
#pragma unroll
    for (int kt = 0; kt < 32; ++kt) {
        const int k0 = kt * 32 + kb;
        AFrag a0, a1;
        a0.h[0] = *(const v8h*)(hrow0 + k0);
        a0.h[1] = *(const v8h*)(hrow0 + k0 + 16);
        a1.h[0] = *(const v8h*)(hrow1 + k0);
        a1.h[1] = *(const v8h*)(hrow1 + k0 + 16);
        const _Float16* bp = Wpd + ((size_t)(kt * 256 + nt0)) * 512 + lane * 16;
        v16h b0 = *(const v16h*)(bp);
        v16h b1 = *(const v16h*)(bp + 512);
        acc00 = __builtin_amdgcn_wmma_f32_16x16x32_f16(false, a0.v, false, b0, (short)0, acc00, false, false);
        acc01 = __builtin_amdgcn_wmma_f32_16x16x32_f16(false, a0.v, false, b1, (short)0, acc01, false, false);
        acc10 = __builtin_amdgcn_wmma_f32_16x16x32_f16(false, a1.v, false, b0, (short)0, acc10, false, false);
        acc11 = __builtin_amdgcn_wmma_f32_16x16x32_f16(false, a1.v, false, b1, (short)0, acc11, false, false);
    }

    const int ncol = lane & 15;
    const int mofs = (lane >> 4) << 3;
    float* g00 = Gd + (size_t)(mt0 * 16 + mofs) * G8_ + (nt0 * 16 + ncol);
    float* g10 = Gd + (size_t)((mt0 + 1) * 16 + mofs) * G8_ + (nt0 * 16 + ncol);
#pragma unroll
    for (int v = 0; v < 8; ++v) {
        g00[(size_t)v * G8_]      = acc00[v];
        g00[(size_t)v * G8_ + 16] = acc01[v];
        g10[(size_t)v * G8_]      = acc10[v];
        g10[(size_t)v * G8_ + 16] = acc11[v];
    }
}

// LSTM cell + rank-1 x*Wih term + fc reduction. One block per batch row.
__global__ void dec_point_kernel(const float* __restrict__ Gd, const float* __restrict__ bcd,
                                 const float* __restrict__ Wihd, float* __restrict__ cd,
                                 _Float16* __restrict__ hd16, float* __restrict__ xprev,
                                 const float* __restrict__ fcW, const float* __restrict__ fcb,
                                 float* __restrict__ out, int t) {
    __shared__ float red[256];
    const int b = blockIdx.x;
    const int tid = threadIdx.x;
    const float x = xprev[b];
    const float* g = Gd + (size_t)b * G8_;
    float sum = 0.0f;
#pragma unroll
    for (int it = 0; it < 4; ++it) {
        int j = tid + it * 256;
        float gi = g[j]            + bcd[j]            + x * Wihd[j];
        float gf = g[H2_ + j]      + bcd[H2_ + j]      + x * Wihd[H2_ + j];
        float gg = g[2 * H2_ + j]  + bcd[2 * H2_ + j]  + x * Wihd[2 * H2_ + j];
        float go = g[3 * H2_ + j]  + bcd[3 * H2_ + j]  + x * Wihd[3 * H2_ + j];
        float cn = sigmoidf_(gf) * cd[(size_t)b * H2_ + j] + sigmoidf_(gi) * tanhf(gg);
        cd[(size_t)b * H2_ + j] = cn;
        float hn = sigmoidf_(go) * tanhf(cn);
        hd16[(size_t)b * H2_ + j] = (_Float16)hn;
        sum += hn * fcW[j];
    }
    red[tid] = sum;
    __syncthreads();
    for (int s = 128; s > 0; s >>= 1) {
        if (tid < s) red[tid] += red[tid + s];
        __syncthreads();
    }
    if (tid == 0) {
        float p = red[0] + fcb[0];
        out[(size_t)b * L_ + t] = p;
        xprev[b] = p;
    }
}

// ---------------------------------------------------------------------------
// Host orchestration
// ---------------------------------------------------------------------------
extern "C" void kernel_launch(void* const* d_in, const int* in_sizes, int n_in,
                              void* d_out, int out_size, void* d_ws, size_t ws_size,
                              hipStream_t stream) {
    const float* input_seq = (const float*)d_in[0];
    const float* eWihf = (const float*)d_in[1];
    const float* eWhhf = (const float*)d_in[2];
    const float* ebihf = (const float*)d_in[3];
    const float* ebhhf = (const float*)d_in[4];
    const float* eWihb = (const float*)d_in[5];
    const float* eWhhb = (const float*)d_in[6];
    const float* ebihb = (const float*)d_in[7];
    const float* ebhhb = (const float*)d_in[8];
    const float* dWih  = (const float*)d_in[9];
    const float* dWhh  = (const float*)d_in[10];
    const float* dbih  = (const float*)d_in[11];
    const float* dbhh  = (const float*)d_in[12];
    const float* fcW   = (const float*)d_in[13];
    const float* fcb   = (const float*)d_in[14];
    float* out = (float*)d_out;

    // workspace carve-up (~35 MB total), 256B aligned
    char* p = (char*)d_ws;
    auto alloc = [&](size_t bytes) -> void* {
        void* r = (void*)p;
        p += (bytes + 255) & ~(size_t)255;
        return r;
    };
    _Float16* X16  = (_Float16*)alloc((size_t)T_ * B_ * F_ * 2);
    _Float16* Wpf  = (_Float16*)alloc((size_t)128 * 20 * 512 * 2);
    _Float16* Wpb  = (_Float16*)alloc((size_t)128 * 20 * 512 * 2);
    _Float16* Wpd  = (_Float16*)alloc((size_t)256 * 32 * 512 * 2);
    float* bcf = (float*)alloc(G4_ * 4);
    float* bcb = (float*)alloc(G4_ * 4);
    float* bcd = (float*)alloc(G8_ * 4);
    float* Gf  = (float*)alloc((size_t)B_ * G4_ * 4);
    float* Gb  = (float*)alloc((size_t)B_ * G4_ * 4);
    float* Gd  = (float*)alloc((size_t)B_ * G8_ * 4);
    float* hf  = (float*)alloc((size_t)B_ * H_ * 4);
    float* cf  = (float*)alloc((size_t)B_ * H_ * 4);
    float* hb  = (float*)alloc((size_t)B_ * H_ * 4);
    float* cb  = (float*)alloc((size_t)B_ * H_ * 4);
    _Float16* h16f = (_Float16*)alloc((size_t)B_ * H_ * 2);
    _Float16* h16b = (_Float16*)alloc((size_t)B_ * H_ * 2);
    float* cd  = (float*)alloc((size_t)B_ * H2_ * 4);
    _Float16* hd16 = (_Float16*)alloc((size_t)B_ * H2_ * 2);
    float* xprev = (float*)alloc((size_t)B_ * 4);

    // zero recurrent state (capture-legal memset nodes)
    hipMemsetAsync(hf, 0, (size_t)B_ * H_ * 4, stream);
    hipMemsetAsync(cf, 0, (size_t)B_ * H_ * 4, stream);
    hipMemsetAsync(hb, 0, (size_t)B_ * H_ * 4, stream);
    hipMemsetAsync(cb, 0, (size_t)B_ * H_ * 4, stream);
    hipMemsetAsync(h16f, 0, (size_t)B_ * H_ * 2, stream);
    hipMemsetAsync(h16b, 0, (size_t)B_ * H_ * 2, stream);
    hipMemsetAsync(xprev, 0, (size_t)B_ * 4, stream);

    // prep
    xcvt_kernel<<<(T_ * B_ * F_ + 255) / 256, 256, 0, stream>>>(input_seq, X16);
    pack_w_kernel<<<(128 * 20 * 512) / 256, 256, 0, stream>>>(eWhhf, eWihf, Wpf, 128, 20, H_, F_);
    pack_w_kernel<<<(128 * 20 * 512) / 256, 256, 0, stream>>>(eWhhb, eWihb, Wpb, 128, 20, H_, F_);
    pack_w_kernel<<<(256 * 32 * 512) / 256, 256, 0, stream>>>(dWhh, dWhh, Wpd, 256, 32, H2_, 1);
    bias_comb_kernel<<<(2 * G4_ + G8_) / 256, 256, 0, stream>>>(ebihf, ebhhf, ebihb, ebhhb,
                                                                dbih, dbhh, bcf, bcb, bcd);

    // bidirectional encoder (serialized recurrence)
    for (int t = 0; t < T_; ++t) {
        enc_gemm_kernel<<<dim3(64, 2), 256, 0, stream>>>(h16f, h16b, X16, Wpf, Wpb, Gf, Gb, t);
        enc_point_kernel<<<(2 * B_ * H_) / 256, 256, 0, stream>>>(Gf, Gb, bcf, bcb,
                                                                  hf, cf, h16f, hb, cb, h16b);
    }

    // decoder init: hd = [hf|hb], cd = [cf|cb]
    dec_init_kernel<<<(B_ * H2_) / 256, 256, 0, stream>>>(hf, hb, cf, cb, hd16, cd);

    // autoregressive decoder
    for (int t = 0; t < L_; ++t) {
        dec_gemm_kernel<<<128, 256, 0, stream>>>(hd16, Wpd, Gd);
        dec_point_kernel<<<B_, 256, 0, stream>>>(Gd, bcd, dWih, cd, hd16, xprev, fcW, fcb, out, t);
    }
}
